// HiResPrecipNet_up_CNN_GNN_1563368096256
// MI455X (gfx1250) — matrix operand order repaired and checked by reading, not compiled, not run on hardware
//
#include <hip/hip_runtime.h>
#include <hip/hip_bf16.h>
#include <stdint.h>

#define TPB 256

typedef __attribute__((ext_vector_type(16))) _Float16 v16h;
typedef __attribute__((ext_vector_type(8)))  float    v8f;

static inline int cdiv_ll(long long a, int b) { return (int)((a + (long long)b - 1) / b); }

// ------------------------------------------------------------------
// utility kernels
// ------------------------------------------------------------------
__global__ void k_fill(float* p, long long n, float v) {
  long long i = (long long)blockIdx.x * blockDim.x + threadIdx.x;
  if (i < n) p[i] = v;
}

__global__ void k_relu(float* p, long long n) {
  long long i = (long long)blockIdx.x * blockDim.x + threadIdx.x;
  if (i < n) p[i] = fmaxf(p[i], 0.f);
}

// ------------------------------------------------------------------
// f16 staging in WMMA *fragment order*.
// A fragment: blk = tm*KC + kc; element i = (blk*32 + lane)*16 + j
//   lane<16: row M=tm*16+lane, halves j map K = hi*16 + j (contiguous, as B)
// B fragment: blk = tn*KC + kc; lane -> column N, halves j -> K sequential
// ------------------------------------------------------------------
__global__ void k_cvtA_frag(const float* __restrict__ A, _Float16* __restrict__ O,
                            int M, int K, int KC, long long n) {
  long long i = (long long)blockIdx.x * blockDim.x + threadIdx.x;
  if (i >= n) return;
  int j    = (int)(i & 15);
  int lane = (int)((i >> 4) & 31);
  long long blk = i >> 9;
  int kc = (int)(blk % KC);
  int tm = (int)(blk / KC);
  int lm = lane & 15, hi = lane >> 4;
  int pr = j >> 1;
  int kk = hi * 16 + j + (pr & 0);
  int m = tm * 16 + lm;
  int k = kc * 32 + kk;
  O[i] = (m < M && k < K) ? (_Float16)A[(long long)m * K + k] : (_Float16)0.f;
}

__global__ void k_cvtB_frag(const float* __restrict__ B, _Float16* __restrict__ O,
                            int K, int N, int KC, long long n) {
  long long i = (long long)blockIdx.x * blockDim.x + threadIdx.x;
  if (i >= n) return;
  int j    = (int)(i & 15);
  int lane = (int)((i >> 4) & 31);
  long long blk = i >> 9;
  int kc = (int)(blk % KC);
  int tn = (int)(blk / KC);
  int lm = lane & 15, hi = lane >> 4;
  int k = kc * 32 + hi * 16 + j;
  int c = tn * 16 + lm;
  O[i] = (k < K) ? (_Float16)B[(long long)k * N + c] : (_Float16)0.f;
}

// ------------------------------------------------------------------
// WMMA GEMM: C[M,N] = A @ B + bias[N]; fragments pre-permuted, so the
// inner loop is 2x global_load_b128 (per operand) + v_wmma_f32_16x16x32_f16.
// 4 waves per block, one 16x16 tile per wave (tile guard is wave-uniform
// so EXEC stays all-ones for WMMA).
// ------------------------------------------------------------------
__global__ void k_wmma_gemm(const _Float16* __restrict__ A, const _Float16* __restrict__ B,
                            const float* __restrict__ bias, float* __restrict__ C,
                            int KC, int tilesN, int tiles, int N, int M) {
  int wave = threadIdx.x >> 5;
  int tile = blockIdx.x * (blockDim.x >> 5) + wave;
  if (tile >= tiles) return;
  int tm = tile / tilesN;
  int tn = tile % tilesN;
  int lane = threadIdx.x & 31;
  const v16h* Af = (const v16h*)A + ((size_t)tm * KC) * 32 + lane;
  const v16h* Bf = (const v16h*)B + ((size_t)tn * KC) * 32 + lane;
  v8f acc = {};
  for (int kc = 0; kc < KC; ++kc) {
    v16h a = Af[(size_t)kc * 32];
    v16h b = Bf[(size_t)kc * 32];
    acc = __builtin_amdgcn_wmma_f32_16x16x32_f16(false, a, false, b, (short)0, acc, false, false);
  }
  int lm = lane & 15, hi = lane >> 4;
  int col = tn * 16 + lm;
  float bb = bias[col];
  int row0 = tm * 16 + 8 * hi;
  if (row0 + 7 < M) {
#pragma unroll
    for (int v = 0; v < 8; ++v)
      C[(size_t)(row0 + v) * N + col] = acc[v] + bb;
  } else {
#pragma unroll
    for (int v = 0; v < 8; ++v)
      if (row0 + v < M) C[(size_t)(row0 + v) * N + col] = acc[v] + bb;
  }
}

// ------------------------------------------------------------------
// GATv2 edge kernels
// ------------------------------------------------------------------
__device__ inline void atomicMaxF(float* a, float v) {
  int cur = __float_as_int(*a);
  while (v > __int_as_float(cur)) {
    int prev = atomicCAS((int*)a, cur, __float_as_int(v));
    if (prev == cur) break;
    cur = prev;
  }
}

__global__ void k_edge_score(const float* __restrict__ xl, const float* __restrict__ xr,
                             const int* __restrict__ src, const int* __restrict__ dst,
                             const float* __restrict__ att, float* __restrict__ sc,
                             float* __restrict__ mm, int E, int H, int C) {
  long long i = (long long)blockIdx.x * blockDim.x + threadIdx.x;
  if (i >= (long long)E * H) return;
  int e = (int)(i / H), h = (int)(i % H);
  int s = src[e], d = dst[e];
  const float4* pl = (const float4*)(xl + (size_t)s * H * C + (size_t)h * C);
  const float4* pr = (const float4*)(xr + (size_t)d * H * C + (size_t)h * C);
  const float4* pa = (const float4*)(att + (size_t)h * C);
  float score = 0.f;
  int C4 = C >> 2;
  for (int c = 0; c < C4; ++c) {
    float4 l = pl[c], r = pr[c], a = pa[c];
    float v0 = l.x + r.x; v0 = (v0 > 0.f) ? v0 : 0.2f * v0;
    float v1 = l.y + r.y; v1 = (v1 > 0.f) ? v1 : 0.2f * v1;
    float v2 = l.z + r.z; v2 = (v2 > 0.f) ? v2 : 0.2f * v2;
    float v3 = l.w + r.w; v3 = (v3 > 0.f) ? v3 : 0.2f * v3;
    score += v0 * a.x + v1 * a.y + v2 * a.z + v3 * a.w;
  }
  sc[i] = score;
  atomicMaxF(&mm[(size_t)d * H + h], score);
}

__global__ void k_edge_exp(float* __restrict__ sc, const int* __restrict__ dst,
                           const float* __restrict__ mm, float* __restrict__ den,
                           float* __restrict__ cnt, int E, int H) {
  long long i = (long long)blockIdx.x * blockDim.x + threadIdx.x;
  if (i >= (long long)E * H) return;
  int e = (int)(i / H), h = (int)(i % H);
  int d = dst[e];
  float ex = expf(sc[i] - mm[(size_t)d * H + h]);
  sc[i] = ex;
  atomicAdd(&den[(size_t)d * H + h], ex);
  if (h == 0) atomicAdd(&cnt[d], 1.f);
}

__global__ void k_edge_scatter(const float* __restrict__ xl, const float* __restrict__ sc,
                               const int* __restrict__ src, const int* __restrict__ dst,
                               const float* __restrict__ den, float* __restrict__ out,
                               int E, int H, int C) {
  long long i = (long long)blockIdx.x * blockDim.x + threadIdx.x;
  long long tot = (long long)E * H * C;
  if (i >= tot) return;
  int HC = H * C;
  int e = (int)(i / HC);
  int j = (int)(i % HC);
  int h = j / C;
  int s = src[e], d = dst[e];
  float alpha = sc[(size_t)e * H + h] / (den[(size_t)d * H + h] + 1e-16f);
  atomicAdd(&out[(size_t)d * HC + j], alpha * xl[(size_t)s * HC + j]);
}

__global__ void k_gat_final(float* __restrict__ out, const float* __restrict__ cnt,
                            const float* __restrict__ bias, int Nd, int HC) {
  long long i = (long long)blockIdx.x * blockDim.x + threadIdx.x;
  if (i >= (long long)Nd * HC) return;
  int n = (int)(i / HC);
  int j = (int)(i % HC);
  out[i] = out[i] / fmaxf(cnt[n], 1.f) + bias[j];
}

// ------------------------------------------------------------------
// BatchNorm over nodes (axes=(0,)); F <= 128; st[f]=mean, st[128+f]=var
// ------------------------------------------------------------------
__global__ void k_bn_stats(const float* __restrict__ x, float* __restrict__ st, int N, int F) {
  int f = blockIdx.x;
  __shared__ float ss[TPB], sq[TPB];
  float s = 0.f, q = 0.f;
  for (int n = threadIdx.x; n < N; n += TPB) {
    float v = x[(long long)n * F + f];
    s += v; q += v * v;
  }
  ss[threadIdx.x] = s; sq[threadIdx.x] = q;
  __syncthreads();
  for (int o = TPB / 2; o > 0; o >>= 1) {
    if ((int)threadIdx.x < o) { ss[threadIdx.x] += ss[threadIdx.x + o]; sq[threadIdx.x] += sq[threadIdx.x + o]; }
    __syncthreads();
  }
  if (threadIdx.x == 0) {
    float mu = ss[0] / (float)N;
    st[f] = mu;
    st[128 + f] = sq[0] / (float)N - mu * mu;
  }
}

__global__ void k_bn_relu(float* __restrict__ x, const float* __restrict__ st,
                          const float* __restrict__ g, const float* __restrict__ b,
                          long long n, int F) {
  long long i = (long long)blockIdx.x * blockDim.x + threadIdx.x;
  if (i >= n) return;
  int f = (int)(i % F);
  float v = (x[i] - st[f]) * rsqrtf(st[128 + f] + 1e-5f) * g[f] + b[f];
  x[i] = fmaxf(v, 0.f);
}

// ------------------------------------------------------------------
// CNN encoder kernels (x: [N,5,5,5] NCHW)
// ------------------------------------------------------------------
__global__ void k_dwconv(const float* __restrict__ x, const float* __restrict__ wt,
                         const float* __restrict__ bs, float* __restrict__ y, int N) {
  long long i = (long long)blockIdx.x * blockDim.x + threadIdx.x;
  if (i >= (long long)N * 125) return;
  int n = (int)(i / 125);
  int r = (int)(i % 125);
  int c = r / 25;
  int p = r % 25;
  int ii = p / 5, jj = p % 5;
  float s = bs[c];
  const float* base = x + (long long)n * 125 + c * 25;
  const float* wk = wt + c * 9;
  for (int di = -1; di <= 1; ++di) {
    int yy = ii + di;
    if (yy < 0 || yy >= 5) continue;
    for (int dj = -1; dj <= 1; ++dj) {
      int xx = jj + dj;
      if (xx < 0 || xx >= 5) continue;
      s += base[yy * 5 + xx] * wk[(di + 1) * 3 + (dj + 1)];
    }
  }
  y[i] = s;
}

__global__ void k_cnn_stats(const float* __restrict__ x, float* __restrict__ st, int N) {
  int c = blockIdx.x;  // 0..4
  __shared__ float ss[TPB], sq[TPB];
  float s = 0.f, q = 0.f;
  int tot = N * 25;
  for (int idx = threadIdx.x; idx < tot; idx += TPB) {
    int n = idx / 25, p = idx % 25;
    float v = x[(long long)n * 125 + c * 25 + p];
    s += v; q += v * v;
  }
  ss[threadIdx.x] = s; sq[threadIdx.x] = q;
  __syncthreads();
  for (int o = TPB / 2; o > 0; o >>= 1) {
    if ((int)threadIdx.x < o) { ss[threadIdx.x] += ss[threadIdx.x + o]; sq[threadIdx.x] += sq[threadIdx.x + o]; }
    __syncthreads();
  }
  if (threadIdx.x == 0) {
    float mu = ss[0] / (float)tot;
    st[c] = mu;
    st[128 + c] = sq[0] / (float)tot - mu * mu;
  }
}

__global__ void k_cnn_bnrelu(float* __restrict__ x, const float* __restrict__ st,
                             const float* __restrict__ g, const float* __restrict__ b, int N) {
  long long i = (long long)blockIdx.x * blockDim.x + threadIdx.x;
  if (i >= (long long)N * 125) return;
  int c = (int)((i / 25) % 5);
  float v = (x[i] - st[c]) * rsqrtf(st[128 + c] + 1e-5f) * g[c] + b[c];
  x[i] = fmaxf(v, 0.f);
}

// maxpool 2x2 stride 2, padding (1,1) with -inf : [N,5,5,5] -> [N,45]
__global__ void k_maxpool(const float* __restrict__ x, float* __restrict__ y, int N) {
  long long i = (long long)blockIdx.x * blockDim.x + threadIdx.x;
  if (i >= (long long)N * 45) return;
  int n = (int)(i / 45);
  int r = (int)(i % 45);
  int c = r / 9;
  int oi = (r % 9) / 3;
  int oj = r % 3;
  float m = -1e30f;
  for (int di = 0; di < 2; ++di) {
    int ii = 2 * oi - 1 + di;
    if (ii < 0 || ii >= 5) continue;
    for (int dj = 0; dj < 2; ++dj) {
      int jj = 2 * oj - 1 + dj;
      if (jj < 0 || jj >= 5) continue;
      m = fmaxf(m, x[(long long)n * 125 + c * 25 + ii * 5 + jj]);
    }
  }
  y[i] = m;
}

// concat: out[n,0]=z[n]; out[n,1..32]=h[n,0..31]
__global__ void k_concat_high(const float* __restrict__ z, const float* __restrict__ h,
                              float* __restrict__ out, int N) {
  long long i = (long long)blockIdx.x * blockDim.x + threadIdx.x;
  if (i >= (long long)N * 33) return;
  int n = (int)(i / 33);
  int j = (int)(i % 33);
  out[i] = (j == 0) ? z[n] : h[(long long)n * 32 + (j - 1)];
}

// small dense layer (pred head)
__global__ void k_linear(const float* __restrict__ x, const float* __restrict__ w,
                         const float* __restrict__ b, float* __restrict__ y,
                         int N, int Ci, int Co, int doRelu) {
  long long i = (long long)blockIdx.x * blockDim.x + threadIdx.x;
  if (i >= (long long)N * Co) return;
  int n = (int)(i / Co);
  int co = (int)(i % Co);
  float s = b[co];
  const float* xr = x + (long long)n * Ci;
  for (int ci = 0; ci < Ci; ++ci) s += xr[ci] * w[ci * Co + co];
  y[i] = doRelu ? fmaxf(s, 0.f) : s;
}

// ------------------------------------------------------------------
// host-side plumbing
// ------------------------------------------------------------------
struct GatIdx { int wl, bl, wr, br, att, bias; };
struct ProcIdx { GatIdx g[5]; int bng[4], bnb[4]; };
struct NetIdx {
  int convw[3], convb[3], bng[3], bnb[3];
  ProcIdx plow, pup, phigh;
  GatIdx l2u, u2h;
  int predw[3], predb[3];
};

static void fill_insertion(NetIdx& x) {  // dict insertion order
  int p = 9;
  for (int i = 0; i < 3; ++i) { x.convw[i] = p++; x.convb[i] = p++; }
  for (int i = 0; i < 3; ++i) { x.bng[i] = p++; x.bnb[i] = p++; }
  auto fg = [&](GatIdx& g) { g.wl = p++; g.bl = p++; g.wr = p++; g.br = p++; g.att = p++; g.bias = p++; };
  auto fp = [&](ProcIdx& pr) {
    for (int g = 0; g < 5; ++g) fg(pr.g[g]);
    for (int i = 0; i < 4; ++i) { pr.bng[i] = p++; pr.bnb[i] = p++; }
  };
  fp(x.plow); fg(x.l2u); fp(x.pup); fg(x.u2h); fp(x.phigh);
  for (int i = 0; i < 3; ++i) { x.predw[i] = p++; x.predb[i] = p++; }
}

static void fill_sorted(NetIdx& x) {  // jax pytree sorted-key order
  int p = 9;
  for (int i = 0; i < 3; ++i) { x.bnb[i] = p++; x.bng[i] = p++; }
  for (int i = 0; i < 3; ++i) { x.convb[i] = p++; x.convw[i] = p++; }
  auto fg = [&](GatIdx& g) { g.att = p++; g.bias = p++; g.bl = p++; g.br = p++; g.wl = p++; g.wr = p++; };
  auto fp = [&](ProcIdx& pr) {
    for (int i = 0; i < 4; ++i) { pr.bnb[i] = p++; pr.bng[i] = p++; }
    for (int g = 0; g < 5; ++g) fg(pr.g[g]);
  };
  fg(x.l2u); fg(x.u2h);
  for (int i = 0; i < 3; ++i) { x.predb[i] = p++; x.predw[i] = p++; }
  fp(x.phigh); fp(x.plow); fp(x.pup);
}

struct Ws {
  float *BX, *BY, *XL, *XR, *SC, *MM, *DN, *CT, *CA, *CB, *ST;
  _Float16 *A16, *B16;
};

static void matmul_wmma(const float* A, int M, int K, const float* W, const float* bias,
                        int N, float* C, const Ws& w, hipStream_t st) {
  int Kp = (K + 31) & ~31;
  int KC = Kp >> 5;
  int Mp = (M + 15) & ~15;
  int tilesM = Mp >> 4, tilesN = N >> 4;
  int tiles = tilesM * tilesN;
  long long na = (long long)tilesM * KC * 512;
  k_cvtA_frag<<<cdiv_ll(na, TPB), TPB, 0, st>>>(A, w.A16, M, K, KC, na);
  long long nb = (long long)tilesN * KC * 512;
  k_cvtB_frag<<<cdiv_ll(nb, TPB), TPB, 0, st>>>(W, w.B16, K, N, KC, nb);
  k_wmma_gemm<<<cdiv_ll(tiles, 4), 128, 0, st>>>(w.A16, w.B16, bias, C, KC, tilesN, tiles, N, M);
}

static void run_gat(void* const* d_in, const GatIdx& gi,
                    const float* xs, int Ns, int Cs, const float* xd, int Nd, int Cd,
                    const int* src, const int* dst, int E, int H, int C,
                    float* out, const Ws& w, hipStream_t st) {
  auto Fp = [&](int i) { return (const float*)d_in[i]; };
  int HC = H * C;
  matmul_wmma(xs, Ns, Cs, Fp(gi.wl), Fp(gi.bl), HC, w.XL, w, st);
  matmul_wmma(xd, Nd, Cd, Fp(gi.wr), Fp(gi.br), HC, w.XR, w, st);
  long long nh = (long long)Nd * H;
  k_fill<<<cdiv_ll(nh, TPB), TPB, 0, st>>>(w.MM, nh, -1e30f);
  k_fill<<<cdiv_ll(nh, TPB), TPB, 0, st>>>(w.DN, nh, 0.f);
  k_fill<<<cdiv_ll(Nd, TPB), TPB, 0, st>>>(w.CT, Nd, 0.f);
  long long no = (long long)Nd * HC;
  k_fill<<<cdiv_ll(no, TPB), TPB, 0, st>>>(out, no, 0.f);
  long long ne = (long long)E * H;
  k_edge_score<<<cdiv_ll(ne, TPB), TPB, 0, st>>>(w.XL, w.XR, src, dst, Fp(gi.att), w.SC, w.MM, E, H, C);
  k_edge_exp<<<cdiv_ll(ne, TPB), TPB, 0, st>>>(w.SC, dst, w.MM, w.DN, w.CT, E, H);
  long long ns = (long long)E * HC;
  k_edge_scatter<<<cdiv_ll(ns, TPB), TPB, 0, st>>>(w.XL, w.SC, src, dst, w.DN, out, E, H, C);
  k_gat_final<<<cdiv_ll(no, TPB), TPB, 0, st>>>(out, w.CT, Fp(gi.bias), Nd, HC);
}

static const float* run_processor(void* const* d_in, const ProcIdx& pi,
                                  float* bufA, float* bufB, const float* x0,
                                  int N, int Cin, int hid,
                                  const int* src, const int* dst, int E,
                                  const Ws& w, hipStream_t st) {
  auto Fp = [&](int i) { return (const float*)d_in[i]; };
  const float* cur = x0;
  for (int g = 0; g < 5; ++g) {
    int H = (g == 4) ? 1 : 2;
    int Cs = (g == 0) ? Cin : 2 * hid;
    float* dbuf = (cur == bufA) ? bufB : bufA;
    run_gat(d_in, pi.g[g], cur, N, Cs, cur, N, Cs, src, dst, E, H, hid, dbuf, w, st);
    long long tot = (long long)N * H * hid;
    if (g < 4) {
      k_bn_stats<<<2 * hid, TPB, 0, st>>>(dbuf, w.ST, N, 2 * hid);
      k_bn_relu<<<cdiv_ll(tot, TPB), TPB, 0, st>>>(dbuf, w.ST, Fp(pi.bng[g]), Fp(pi.bnb[g]), tot, 2 * hid);
    } else {
      k_relu<<<cdiv_ll(tot, TPB), TPB, 0, st>>>(dbuf, tot);
    }
    cur = dbuf;
  }
  return cur;
}

extern "C" void kernel_launch(void* const* d_in, const int* in_sizes, int n_in,
                              void* d_out, int out_size, void* d_ws, size_t ws_size,
                              hipStream_t stream) {
  (void)out_size; (void)ws_size;
  const int N_LOW = 10000, N_UP = 30000, N_HIGH = 60000;
  const int E_LOW = 160000, E_L2U = 120000, E_UP = 300000, E_U2H = 240000, E_HIGH = 480000;

  NetIdx ix;
  if (n_in > 9 && in_sizes[9] == 5) fill_sorted(ix); else fill_insertion(ix);

  auto Fp = [&](int i) { return (const float*)d_in[i]; };
  auto Ip = [&](int i) { return (const int*)d_in[i]; };

  float* base = (float*)d_ws;
  Ws w;
  w.BX = base; base += 4000000;   // node feature ping (<= 60000*64 or 10000*128)
  w.BY = base; base += 4000000;   // node feature pong
  w.XL = base; base += 4000000;   // GAT xl
  w.XR = base; base += 4000000;   // GAT xr
  w.SC = base; base += 1000000;   // edge scores / exp (<= 480000*2)
  w.MM = base; base += 130000;    // segment max
  w.DN = base; base += 130000;    // segment denom
  w.CT = base; base += 65000;     // in-degree counts
  w.CA = base; base += 1250000;   // cnn tmp A
  w.CB = base; base += 1250000;   // cnn tmp B
  w.ST = base; base += 512;       // bn stats (mean[0..127], var[128..255])
  w.A16 = (_Float16*)base; base += 2050000;  // 4.1M halves (32B aligned by construction)
  w.B16 = (_Float16*)base;                   // 33K halves

  // ---------------- CNN encoder ----------------
  {
    long long tot = (long long)N_LOW * 125;
    int gb = cdiv_ll(tot, TPB);
    k_dwconv<<<gb, TPB, 0, stream>>>(Fp(0), Fp(ix.convw[0]), Fp(ix.convb[0]), w.CA, N_LOW);
    k_cnn_stats<<<5, TPB, 0, stream>>>(w.CA, w.ST, N_LOW);
    k_cnn_bnrelu<<<gb, TPB, 0, stream>>>(w.CA, w.ST, Fp(ix.bng[0]), Fp(ix.bnb[0]), N_LOW);
    k_dwconv<<<gb, TPB, 0, stream>>>(w.CA, Fp(ix.convw[1]), Fp(ix.convb[1]), w.CB, N_LOW);
    k_cnn_stats<<<5, TPB, 0, stream>>>(w.CB, w.ST, N_LOW);
    k_cnn_bnrelu<<<gb, TPB, 0, stream>>>(w.CB, w.ST, Fp(ix.bng[1]), Fp(ix.bnb[1]), N_LOW);
    k_dwconv<<<gb, TPB, 0, stream>>>(w.CB, Fp(ix.convw[2]), Fp(ix.convb[2]), w.CA, N_LOW);
    k_cnn_stats<<<5, TPB, 0, stream>>>(w.CA, w.ST, N_LOW);
    k_cnn_bnrelu<<<gb, TPB, 0, stream>>>(w.CA, w.ST, Fp(ix.bng[2]), Fp(ix.bnb[2]), N_LOW);
    long long tp = (long long)N_LOW * 45;
    k_maxpool<<<cdiv_ll(tp, TPB), TPB, 0, stream>>>(w.CA, w.BX, N_LOW);
  }

  // ---------------- proc_low ----------------
  const float* hlow = run_processor(d_in, ix.plow, w.BX, w.BY, w.BX, N_LOW, 45, 64,
                                    Ip(4), Ip(4) + E_LOW, E_LOW, w, stream);

  // ---------------- down low -> up ----------------
  float* oup = (hlow == w.BX) ? w.BY : w.BX;
  run_gat(d_in, ix.l2u, hlow, N_LOW, 64, Fp(1), N_UP, 1,
          Ip(5), Ip(5) + E_L2U, E_L2U, 1, 32, oup, w, stream);

  // ---------------- proc_up ----------------
  const float* hup = run_processor(d_in, ix.pup, w.BX, w.BY, oup, N_UP, 32, 32,
                                   Ip(6), Ip(6) + E_UP, E_UP, w, stream);

  // ---------------- down up -> high ----------------
  float* ohigh = (hup == w.BX) ? w.BY : w.BX;
  run_gat(d_in, ix.u2h, hup, N_UP, 32, Fp(2), N_HIGH, 1,
          Ip(7), Ip(7) + E_U2H, E_U2H, 1, 32, ohigh, w, stream);

  // ---------------- concat z_std_high ----------------
  float* cat = (ohigh == w.BX) ? w.BY : w.BX;
  k_concat_high<<<cdiv_ll((long long)N_HIGH * 33, TPB), TPB, 0, stream>>>(Fp(3), ohigh, cat, N_HIGH);

  // ---------------- proc_high ----------------
  const float* hhigh = run_processor(d_in, ix.phigh, w.BX, w.BY, cat, N_HIGH, 33, 32,
                                     Ip(8), Ip(8) + E_HIGH, E_HIGH, w, stream);

  // ---------------- prediction head ----------------
  float* t1 = (hhigh == w.BX) ? w.BY : w.BX;
  k_linear<<<cdiv_ll((long long)N_HIGH * 32, TPB), TPB, 0, stream>>>(
      hhigh, Fp(ix.predw[0]), Fp(ix.predb[0]), t1, N_HIGH, 32, 32, 1);
  float* t2 = (t1 == w.BX) ? w.BY : w.BX;
  k_linear<<<cdiv_ll((long long)N_HIGH * 16, TPB), TPB, 0, stream>>>(
      t1, Fp(ix.predw[1]), Fp(ix.predb[1]), t2, N_HIGH, 32, 16, 1);
  k_linear<<<cdiv_ll((long long)N_HIGH, TPB), TPB, 0, stream>>>(
      t2, Fp(ix.predw[2]), Fp(ix.predb[2]), (float*)d_out, N_HIGH, 16, 1, 0);
}
